// TripletLoss_36627481100926
// MI455X (gfx1250) — matrix operand order repaired and checked by reading, not compiled, not run on hardware
//
#include <hip/hip_runtime.h>

typedef __attribute__((ext_vector_type(16))) _Float16 v16h;
typedef __attribute__((ext_vector_type(8)))  float    v8f;

#define C_DIM 8192
#define D_DIM 512
#define BIGF  3.0e38f

union HF16 { v16h h; uint4 u[2]; };

// A-matrix 16x32 f16 fragment (ISA 05_wmma layout):
// lanes 0-15 (h=0): VGPR0-3 = K k0+0..7, VGPR4-7 = K k0+16..23, row = M = lane
// lanes 16-31 (h=1): VGPR0-3 = K k0+8..15, VGPR4-7 = K k0+24..31
__device__ __forceinline__ v16h load_frag_A(const _Float16* __restrict__ G, int row, int k0, int h) {
  HF16 f;
  const _Float16* base = G + (size_t)row * D_DIM;
  f.u[0] = *(const uint4*)(base + k0 + h * 8);
  f.u[1] = *(const uint4*)(base + k0 + 16 + h * 8);
  return f.h;
}

// B-matrix 32x16 f16 fragment: lane n (0-15) = column n; same K order as A:
// h=0 -> K k0+0..7, k0+16..23; h=1 -> K k0+8..15, k0+24..31.
// B[k][n] = G[j0+n][k]  (so B column n is row j0+n of G)
__device__ __forceinline__ v16h load_frag_B(const _Float16* __restrict__ G, int row, int k0, int h) {
  HF16 f;
  const _Float16* p = G + (size_t)row * D_DIM + k0 + h * 8;
  f.u[0] = *(const uint4*)(p);
  f.u[1] = *(const uint4*)(p + 16);
  return f.h;
}

// -------- Phase 0: L2-normalize rows, fp32 -> f16 -----------------------
__global__ __launch_bounds__(256) void tl_normalize_kernel(
    const float* __restrict__ X, _Float16* __restrict__ G) {
  const int lane = threadIdx.x & 31;
  const int wave = threadIdx.x >> 5;
  const int row  = blockIdx.x * 8 + wave;
  const float4* src = (const float4*)(X + (size_t)row * D_DIM + lane * 16);
  float4 x[4];
  float ss = 0.f;
#pragma unroll
  for (int t = 0; t < 4; ++t) {
    x[t] = src[t];
    ss += x[t].x * x[t].x + x[t].y * x[t].y + x[t].z * x[t].z + x[t].w * x[t].w;
  }
#pragma unroll
  for (int off = 1; off <= 16; off <<= 1) ss += __shfl_xor(ss, off, 32);
  const float s = 1.0f / sqrtf(ss);
  union { _Float16 hh[16]; uint4 uu[2]; } o;
  const float* xf = (const float*)x;
#pragma unroll
  for (int q = 0; q < 16; ++q) o.hh[q] = (_Float16)(xf[q] * s);
  uint4* dst = (uint4*)(G + (size_t)row * D_DIM + lane * 16);
  dst[0] = o.uu[0];
  dst[1] = o.uu[1];
}

// -------- Phase 1: fused S = G*G^T + online row stats -------------------
// Block = one 16-row M-tile. Each of the 8 waves sweeps a 1024-column
// slice (64 tiles, processed 2 at a time for WMMA ILP); partial online
// states are merged through LDS at the end (ascending-column merge order
// preserves first-index tie-breaking of argmax/argmin).
__global__ __launch_bounds__(256) void tl_rowstats_kernel(
    const _Float16* __restrict__ G, float* __restrict__ margin,
    int* __restrict__ ms, int* __restrict__ ls) {
  const int lane = threadIdx.x & 31;
  const int wave = threadIdx.x >> 5;   // column-split index 0..7
  const int i0   = blockIdx.x * 16;
  const int h    = lane >> 4;          // half-wave
  const int r    = lane & 15;

  // Cache the full 16-row A panel (K=512) in registers: 16 frags x 8 VGPRs.
  v16h A[16];
#pragma unroll
  for (int t = 0; t < 16; ++t) A[t] = load_frag_A(G, i0 + r, t * 32, h);

  // Per-lane online state for 8 rows (this half's rows i0 + h*8 + e),
  // restricted to columns in this wave's slice with col % 16 == r.
  float m[8], z[8], nv[8];
  int   mi[8], ni[8];
#pragma unroll
  for (int e = 0; e < 8; ++e) { m[e] = -BIGF; z[e] = 0.f; nv[e] = BIGF; mi[e] = 0; ni[e] = 0; }

  const int jt_beg = wave * (C_DIM / 8 / 16);   // 64 tiles per wave
  for (int jt = jt_beg; jt < jt_beg + 64; jt += 2) {
    const int j0 = jt * 16;
    v8f acc0 = {0.f, 0.f, 0.f, 0.f, 0.f, 0.f, 0.f, 0.f};
    v8f acc1 = {0.f, 0.f, 0.f, 0.f, 0.f, 0.f, 0.f, 0.f};
#pragma unroll
    for (int t = 0; t < 16; ++t) {
      v16h B0 = load_frag_B(G, j0 + r,      t * 32, h);
      v16h B1 = load_frag_B(G, j0 + 16 + r, t * 32, h);
      acc0 = __builtin_amdgcn_wmma_f32_16x16x32_f16(false, A[t], false, B0, (short)0, acc0, false, false);
      acc1 = __builtin_amdgcn_wmma_f32_16x16x32_f16(false, A[t], false, B1, (short)0, acc1, false, false);
    }
#pragma unroll
    for (int half = 0; half < 2; ++half) {
      const int colg = j0 + half * 16 + r;
#pragma unroll
      for (int e = 0; e < 8; ++e) {
        const int   rowg = i0 + h * 8 + e;
        const float v    = (half == 0) ? acc0[e] : acc1[e];
        const bool  diag = (rowg == colg);
        const float vmax = diag ? -BIGF : v;   // excluded -> exp underflows to 0
        const float vmin = diag ?  BIGF : v;
        // online softmax (running max + rescaled sum of exps)
        const float M = fmaxf(m[e], vmax);
        z[e] = z[e] * __expf(m[e] - M) + __expf(vmax - M);
        if (vmax > m[e]) mi[e] = colg;         // ties keep earlier column
        m[e] = M;
        if (vmin < nv[e]) { nv[e] = vmin; ni[e] = colg; }
      }
    }
  }

  // Cross-lane merge within each 16-lane half (xor 1,2,4,8 stays in-half).
#pragma unroll
  for (int e = 0; e < 8; ++e) {
#pragma unroll
    for (int off = 1; off <= 8; off <<= 1) {
      const float om  = __shfl_xor(m[e],  off, 32);
      const float oz  = __shfl_xor(z[e],  off, 32);
      const int   omi = __shfl_xor(mi[e], off, 32);
      const float onv = __shfl_xor(nv[e], off, 32);
      const int   oni = __shfl_xor(ni[e], off, 32);
      const float M = fmaxf(m[e], om);
      z[e]  = z[e] * __expf(m[e] - M) + oz * __expf(om - M);
      mi[e] = (om > m[e]) ? omi : ((m[e] > om) ? mi[e] : (mi[e] < omi ? mi[e] : omi));
      m[e]  = M;
      ni[e] = (onv < nv[e]) ? oni : ((nv[e] < onv) ? ni[e] : (ni[e] < oni ? ni[e] : oni));
      nv[e] = fminf(nv[e], onv);
    }
  }

  // Publish each wave's per-row partial state to LDS.
  __shared__ float smem[8][16][5];
#pragma unroll
  for (int e = 0; e < 8; ++e) {
    if (r == e) {
      const int row = h * 8 + e;
      smem[wave][row][0] = m[e];
      smem[wave][row][1] = z[e];
      smem[wave][row][2] = __int_as_float(mi[e]);
      smem[wave][row][3] = nv[e];
      smem[wave][row][4] = __int_as_float(ni[e]);
    }
  }
  __syncthreads();

  // Wave 0, lanes 0..15: merge the 8 column-slice states per row
  // (ascending slice order == ascending column ranges -> ties keep the
  // earlier/lower index, matching jnp.argmax/argmin semantics).
  if (wave == 0 && lane < 16) {
    float M  = smem[0][lane][0];
    float Z  = smem[0][lane][1];
    int   MI = __float_as_int(smem[0][lane][2]);
    float NV = smem[0][lane][3];
    int   NI = __float_as_int(smem[0][lane][4]);
#pragma unroll
    for (int s2 = 1; s2 < 8; ++s2) {
      const float om  = smem[s2][lane][0];
      const float oz  = smem[s2][lane][1];
      const int   omi = __float_as_int(smem[s2][lane][2]);
      const float onv = smem[s2][lane][3];
      const int   oni = __float_as_int(smem[s2][lane][4]);
      const float MM = fmaxf(M, om);
      Z  = Z * __expf(M - MM) + oz * __expf(om - MM);
      MI = (om > M) ? omi : ((M > om) ? MI : (MI < omi ? MI : omi));
      M  = MM;
      NI = (onv < NV) ? oni : ((NV < onv) ? NI : (NI < oni ? NI : oni));
      NV = fminf(NV, onv);
    }
    const int row = i0 + lane;
    margin[row] = (1.0f - __expf(NV - M)) / Z;  // P[ms] - P[ls]
    ms[row] = MI;
    ls[row] = NI;
  }
}

// -------- Phase 2: triplet distances per row ----------------------------
__global__ __launch_bounds__(256) void tl_dist_kernel(
    const float* __restrict__ w, const float* __restrict__ margin,
    const int* __restrict__ ms, const int* __restrict__ ls,
    float* __restrict__ terms) {
  const int lane = threadIdx.x & 31;
  const int wave = threadIdx.x >> 5;
  const int i    = blockIdx.x * 8 + wave;
  const int a = ms[i], b = ls[i];
  const float4* wi = (const float4*)(w + (size_t)i * D_DIM);
  const float4* wa = (const float4*)(w + (size_t)a * D_DIM);
  const float4* wb = (const float4*)(w + (size_t)b * D_DIM);
  float da = 0.f, db = 0.f;
#pragma unroll
  for (int t = 0; t < 4; ++t) {
    const int idx = lane + t * 32;
    const float4 vi = wi[idx], va = wa[idx], vb = wb[idx];
    float dx = vi.x - va.x, dy = vi.y - va.y, dzc = vi.z - va.z, dw = vi.w - va.w;
    da += dx * dx + dy * dy + dzc * dzc + dw * dw;
    dx = vi.x - vb.x; dy = vi.y - vb.y; dzc = vi.z - vb.z; dw = vi.w - vb.w;
    db += dx * dx + dy * dy + dzc * dzc + dw * dw;
  }
#pragma unroll
  for (int off = 1; off <= 16; off <<= 1) {
    da += __shfl_xor(da, off, 32);
    db += __shfl_xor(db, off, 32);
  }
  if (lane == 0)
    terms[i] = fmaxf(0.f, sqrtf(da) - sqrtf(db) + margin[i]);
}

// -------- Phase 3: deterministic mean -----------------------------------
__global__ __launch_bounds__(256) void tl_mean_kernel(
    const float* __restrict__ terms, float* __restrict__ out) {
  __shared__ float sm[256];
  float s = 0.f;
  for (int k = threadIdx.x; k < C_DIM; k += 256) s += terms[k];
  sm[threadIdx.x] = s;
  __syncthreads();
  for (int wdt = 128; wdt > 0; wdt >>= 1) {
    if (threadIdx.x < wdt) sm[threadIdx.x] += sm[threadIdx.x + wdt];
    __syncthreads();
  }
  if (threadIdx.x == 0) out[0] = sm[0] * (1.0f / (float)C_DIM);
}

extern "C" void kernel_launch(void* const* d_in, const int* in_sizes, int n_in,
                              void* d_out, int out_size, void* d_ws, size_t ws_size,
                              hipStream_t stream) {
  const float* gt = (const float*)d_in[0];  // [8192,512] fp32
  const float* w  = (const float*)d_in[1];  // [8192,512] fp32

  char* ws = (char*)d_ws;
  _Float16* G   = (_Float16*)ws;                              // 8 MB
  size_t off = (size_t)C_DIM * D_DIM * sizeof(_Float16);
  float* margin = (float*)(ws + off);            off += C_DIM * sizeof(float);
  int*   ms     = (int*)(ws + off);              off += C_DIM * sizeof(int);
  int*   ls     = (int*)(ws + off);              off += C_DIM * sizeof(int);
  float* terms  = (float*)(ws + off);

  tl_normalize_kernel<<<C_DIM / 8, 256, 0, stream>>>(gt, G);
  tl_rowstats_kernel<<<C_DIM / 16, 256, 0, stream>>>(G, margin, ms, ls);
  tl_dist_kernel<<<C_DIM / 8, 256, 0, stream>>>(w, margin, ms, ls, terms);
  tl_mean_kernel<<<1, 256, 0, stream>>>(terms, (float*)d_out);
}
